// RelativeAttention_16269336117994
// MI455X (gfx1250) — matrix operand order repaired and checked by reading, not compiled, not run on hardware
//
#include <hip/hip_runtime.h>

// ---------------------------------------------------------------------------
// CDNA5 (gfx1250) fused relative-position attention, bf16 WMMA pipeline
// + async global->LDS staging + global prefetch + packed bf16 LDS staging.
// ---------------------------------------------------------------------------

typedef __attribute__((ext_vector_type(16))) __bf16 v16bf;
typedef __attribute__((ext_vector_type(8)))  float  v8f;

union BFrag16 { v16bf v; unsigned short u[16]; };

__device__ __forceinline__ unsigned short f2bf(float f) {
  unsigned int u = __float_as_uint(f);
  unsigned int lsb = (u >> 16) & 1u;
  return (unsigned short)((u + 0x7FFFu + lsb) >> 16);   // round-to-nearest-even
}
__device__ __forceinline__ unsigned int f2bf_pk(float lo, float hi) {
  return (unsigned int)f2bf(lo) | ((unsigned int)f2bf(hi) << 16);
}

// --- gfx1250 async global->LDS path (guarded: falls back if toolchain lacks it)
#if defined(__AMDGCN__) &&                                              \
    __has_builtin(__builtin_amdgcn_global_load_async_to_lds_b64) &&     \
    __has_builtin(__builtin_amdgcn_global_load_async_to_lds_b128) &&    \
    __has_builtin(__builtin_amdgcn_s_wait_asynccnt)
#define USE_ASYNC_LDS 1
#else
#define USE_ASYNC_LDS 0
#endif

// Builtin parameter types (verified via hipcc diagnostics):
//   b128: (v4i AS1*, v4i AS3*, imm int, imm int); b64: v2i likewise.
typedef __attribute__((__vector_size__(4 * sizeof(int)))) int vs4i;
typedef __attribute__((__vector_size__(2 * sizeof(int)))) int vs2i;
typedef __attribute__((address_space(1))) vs4i* gas_v4i;
typedef __attribute__((address_space(3))) vs4i* las_v4i;
typedef __attribute__((address_space(1))) vs2i* gas_v2i;
typedef __attribute__((address_space(3))) vs2i* las_v2i;

__device__ __forceinline__ gas_v4i gas4(const void* p) {
  return (gas_v4i)(unsigned long long)p;                 // flat global == AS1
}
__device__ __forceinline__ las_v4i las4(const void* p) {
  return (las_v4i)(unsigned int)(unsigned long long)p;   // LDS offset low 32b
}
__device__ __forceinline__ gas_v2i gas2(const void* p) {
  return (gas_v2i)(unsigned long long)p;
}
__device__ __forceinline__ las_v2i las2(const void* p) {
  return (las_v2i)(unsigned int)(unsigned long long)p;
}

// B=8, C=256, H=W=32 -> L=1024, NH=8, DK=DV=32, OUT_C=256.

// ---------------------------------------------------------------------------
// Kernel 1: QKV projection.  xf(b,l,c) = x(b,c,l);  Q/K/V = xf @ W (256x256).
// Output: bf16, layout (B, NH, L, 32) row-major.
// grid = (48 = 3 matrices x 16 N-tiles, 8 M-blocks of 128 rows, B)
// ---------------------------------------------------------------------------
__global__ __launch_bounds__(256) void qkv_proj_kernel(
    const float* __restrict__ x,
    const float* __restrict__ Qw, const float* __restrict__ Kw,
    const float* __restrict__ Vw,
    unsigned short* __restrict__ qo, unsigned short* __restrict__ ko,
    unsigned short* __restrict__ vo)
{
  const int b  = blockIdx.z;
  const int l0 = blockIdx.y * 128;
  const int ntile = blockIdx.x;
  const int mat = ntile >> 4;                 // 0=Q 1=K 2=V
  const int n0  = (ntile & 15) * 16;
  const float* W = (mat == 0) ? Qw : (mat == 1) ? Kw : Vw;
  unsigned short* Out = (mat == 0) ? qo : (mat == 1) ? ko : vo;

  // Row pad 34 -> packed u32 stores are 4B aligned; dword-bank m*17+k stays
  // conflict-free across lanes (17 odd).
  __shared__ __attribute__((aligned(16))) unsigned short As[128][34];
  __shared__ __attribute__((aligned(16))) unsigned short Bs[32][16];

  const int tid  = threadIdx.x;
  const int lane = tid & 31;
  const int wid  = tid >> 5;
  const int mbase = wid * 16;
  const int Ml   = lane & 15;
  const int koff = (lane < 16) ? 0 : 8;       // A-operand K split per ISA table

  v8f acc = {};

  for (int k0 = 0; k0 < 256; k0 += 32) {
    // Prefetch next k-slab of x one iteration ahead (WGP scope).
    if (k0 + 32 < 256) {
      __builtin_prefetch(
          &x[((size_t)(b * 256 + k0 + 32 + (tid >> 3)) << 10) + l0 + (tid & 7) * 16],
          0, 3);
    }
    // Stage A-slab: two K-rows per thread -> one packed b32 LDS store.
    for (int it = 0; it < 8; ++it) {
      int lin = it * 256 + tid;               // 0..2047 pairs
      int kp2 = lin >> 7;                     // kk pair index 0..15
      int m   = lin & 127;
      const float* xr = &x[((size_t)(b * 256 + k0 + kp2 * 2) << 10) + l0 + m];
      *(unsigned int*)&As[m][kp2 * 2] = f2bf_pk(xr[0], xr[1024]);
    }
    // Stage B-tile: W[(k0+kk)*256 + n0+n], packed pairs.
    {
      int kk = tid >> 3;                      // 0..31
      int nn = (tid & 7) * 2;                 // 0,2,...,14
      const float* wr = &W[(k0 + kk) * 256 + n0 + nn];
      *(unsigned int*)&Bs[kk][nn] = f2bf_pk(wr[0], wr[1]);
    }
    __syncthreads();

    BFrag16 a, bb;
    #pragma unroll
    for (int j = 0; j < 16; ++j) {
      int kk = koff + ((j < 8) ? j : (j + 8));
      a.u[j] = As[mbase + Ml][kk];
    }
    #pragma unroll
    for (int t = 0; t < 16; ++t) bb.u[t] = Bs[koff + ((t < 8) ? t : (t + 8))][Ml];   // lane = N index, K per ISA table

    acc = __builtin_amdgcn_wmma_f32_16x16x32_bf16(
        false, a.v, false, bb.v, (short)0, acc, false, false);
    __syncthreads();
  }

  // C-layout store: VGPR r -> rows r (lanes 0-15) / r+8 (lanes 16-31).
  const int n    = lane & 15;
  const int mofs = (lane < 16) ? 0 : 8;
  const int o = n0 + n;
  const int h = o >> 5;
  const int d = o & 31;
  #pragma unroll
  for (int r = 0; r < 8; ++r) {
    int l = l0 + mbase + mofs + r;
    Out[((((size_t)(b * 8 + h) << 10) + l) << 5) + d] = f2bf(acc[r]);
  }
}

// ---------------------------------------------------------------------------
// Kernel 2: fused flash attention with relative-position bias.
// One block = 128 query rows of one (b,h); 8 waves x 16-row tiles.
// Q and V tiles are staged with GLOBAL_LOAD_ASYNC_TO_LDS (pure byte moves,
// retired with s_wait_asynccnt); K needs a transpose so it stages manually.
// grid = (8 q-blocks, 64 (b,h) pairs)
// ---------------------------------------------------------------------------
__global__ __launch_bounds__(256) void attn_kernel(
    const unsigned short* __restrict__ q,
    const unsigned short* __restrict__ k,
    const unsigned short* __restrict__ v,
    const float* __restrict__ rel_bias,      // (8, 3969)
    const int*   __restrict__ rel_idx,       // (1024, 1024)
    unsigned short* __restrict__ ao)         // (B, L, 256) bf16
{
  const int qblk = blockIdx.x;
  const int bh   = blockIdx.y;
  const int b = bh >> 3;
  const int h = bh & 7;

  const unsigned short* qp = q + ((size_t)bh << 15);   // *1024*32
  const unsigned short* kp = k + ((size_t)bh << 15);
  const unsigned short* vp = v + ((size_t)bh << 15);
  const float* biasp = rel_bias + h * 3969;

  __shared__ __attribute__((aligned(16))) unsigned short Qs[128][32];
  __shared__ __attribute__((aligned(16))) unsigned short Ks[32][33];   // K^T
  __shared__ __attribute__((aligned(16))) unsigned short Vs[32 * 32];
  __shared__ __attribute__((aligned(16))) unsigned short Ps[8][16][32];

  const int tid  = threadIdx.x;
  const int lane = tid & 31;
  const int wid  = tid >> 5;
  const int mbase = wid * 16;
  const int Ml   = lane & 15;
  const int koff = (lane < 16) ? 0 : 8;
  const int n    = lane & 15;
  const int mofs = (lane < 16) ? 0 : 8;

  // Stage the whole 128x32 Q tile (contiguous 8KB) once.
#if USE_ASYNC_LDS
  {
    const unsigned short* src = qp + (size_t)(qblk * 128) * 32;
    #pragma unroll
    for (int it = 0; it < 2; ++it) {
      __builtin_amdgcn_global_load_async_to_lds_b128(
          gas4(src + it * 2048 + tid * 8),
          las4(&Qs[0][0] + it * 2048 + tid * 8), 0, 0);
    }
    __builtin_amdgcn_s_wait_asynccnt(0);
  }
#else
  {
    const unsigned int* src = (const unsigned int*)(qp + (size_t)(qblk * 128) * 32);
    unsigned int* dst = (unsigned int*)&Qs[0][0];
    #pragma unroll
    for (int it = 0; it < 8; ++it) dst[it * 256 + tid] = src[it * 256 + tid];
  }
#endif
  __syncthreads();

  BFrag16 aQ;
  #pragma unroll
  for (int j = 0; j < 16; ++j) {
    int kk = koff + ((j < 8) ? j : (j + 8));
    aQ.u[j] = Qs[mbase + Ml][kk];
  }

  float m_run[8], l_run[8];
  #pragma unroll
  for (int r = 0; r < 8; ++r) { m_run[r] = -3.0e38f; l_run[r] = 0.0f; }
  v8f acc0 = {}, acc1 = {};

  const int ibase = qblk * 128 + mbase + mofs;   // query row for VGPR r = ibase + r

  for (int j0 = 0; j0 < 1024; j0 += 32) {
    __syncthreads();                              // everyone done with Ks/Vs

    // V tile: pure byte move -> async to LDS (8B per thread covers 2KB).
#if USE_ASYNC_LDS
    __builtin_amdgcn_global_load_async_to_lds_b64(
        gas2(vp + (size_t)j0 * 32 + tid * 4),
        las2(&Vs[tid * 4]), 0, 0);
#endif
    // K tile needs dk<->key transpose: manual staging.
    #pragma unroll
    for (int it = 0; it < 4; ++it) {
      int lin = it * 256 + tid;                   // 0..1023
      int key = lin >> 5;
      int dk  = lin & 31;
      Ks[dk][key] = kp[(size_t)(j0 + key) * 32 + dk];
#if !USE_ASYNC_LDS
      Vs[lin]     = vp[(size_t)j0 * 32 + lin];
#endif
    }
    // Prefetch next iteration's K/V tiles (64B per lane-slot covers 2KB each).
    if (j0 + 32 < 1024 && tid < 32) {
      __builtin_prefetch(kp + (size_t)(j0 + 32) * 32 + tid * 32, 0, 3);
      __builtin_prefetch(vp + (size_t)(j0 + 32) * 32 + tid * 32, 0, 3);
    }
#if USE_ASYNC_LDS
    __builtin_amdgcn_s_wait_asynccnt(0);
#endif
    __syncthreads();

    // QK^T: A = q(16x32), B = k^T(32x16); lane = key column, dk per ISA table.
    BFrag16 bK0, bK1;
    #pragma unroll
    for (int t = 0; t < 16; ++t) {
      int kk = koff + ((t < 8) ? t : (t + 8));
      bK0.u[t] = Ks[kk][Ml];
      bK1.u[t] = Ks[kk][16 + Ml];
    }
    v8f z = {};
    v8f s0 = __builtin_amdgcn_wmma_f32_16x16x32_bf16(
        false, aQ.v, false, bK0.v, (short)0, z, false, false);
    v8f s1 = __builtin_amdgcn_wmma_f32_16x16x32_bf16(
        false, aQ.v, false, bK1.v, (short)0, z, false, false);

    // Bias gather + online softmax (16-lane butterfly matches C-layout rows).
    float p0[8], p1[8];
    #pragma unroll
    for (int r = 0; r < 8; ++r) {
      int i = ibase + r;
      float t0 = s0[r] + biasp[rel_idx[((size_t)i << 10) + j0 + n]];
      float t1 = s1[r] + biasp[rel_idx[((size_t)i << 10) + j0 + 16 + n]];
      float tm = fmaxf(t0, t1);
      #pragma unroll
      for (int msk = 8; msk >= 1; msk >>= 1)
        tm = fmaxf(tm, __shfl_xor(tm, msk, 32));
      float mnew  = fmaxf(m_run[r], tm);
      float alpha = __expf(m_run[r] - mnew);
      m_run[r] = mnew;
      float e0 = __expf(t0 - mnew);
      float e1 = __expf(t1 - mnew);
      float rs = e0 + e1;
      #pragma unroll
      for (int msk = 8; msk >= 1; msk >>= 1)
        rs += __shfl_xor(rs, msk, 32);
      l_run[r] = l_run[r] * alpha + rs;
      acc0[r] *= alpha;
      acc1[r] *= alpha;
      p0[r] = e0; p1[r] = e1;
    }

    // Relayout P: C-layout -> A-layout via per-wave LDS region (in-order DS).
    #pragma unroll
    for (int r = 0; r < 8; ++r) {
      Ps[wid][mofs + r][n]      = f2bf(p0[r]);
      Ps[wid][mofs + r][16 + n] = f2bf(p1[r]);
    }
    BFrag16 aP;
    #pragma unroll
    for (int j = 0; j < 16; ++j) {
      int kk = koff + ((j < 8) ? j : (j + 8));
      aP.u[j] = Ps[wid][Ml][kk];
    }

    // PV: B = v(32keys x 16dv); lane = dv column, key per ISA table.
    BFrag16 bV0, bV1;
    #pragma unroll
    for (int t = 0; t < 16; ++t) {
      int kk = koff + ((t < 8) ? t : (t + 8));
      bV0.u[t] = Vs[kk * 32 + Ml];
      bV1.u[t] = Vs[kk * 32 + 16 + Ml];
    }
    acc0 = __builtin_amdgcn_wmma_f32_16x16x32_bf16(
        false, aP.v, false, bV0.v, (short)0, acc0, false, false);
    acc1 = __builtin_amdgcn_wmma_f32_16x16x32_bf16(
        false, aP.v, false, bV1.v, (short)0, acc1, false, false);
  }

  // Epilogue: normalize and store to (B, L, NH*DV) bf16 at column h*32+d.
  #pragma unroll
  for (int r = 0; r < 8; ++r) {
    float inv = 1.0f / l_run[r];
    int i = ibase + r;
    size_t base = ((size_t)(b * 1024 + i) << 8) + h * 32;
    ao[base + n]      = f2bf(acc0[r] * inv);
    ao[base + 16 + n] = f2bf(acc1[r] * inv);
  }
}

// ---------------------------------------------------------------------------
// Kernel 3: output projection + bias, store fp32 in (B, 256, 32, 32).
// grid = (16 N-tiles, 8 M-blocks, B)
// ---------------------------------------------------------------------------
__global__ __launch_bounds__(256) void ff_kernel(
    const unsigned short* __restrict__ ao,   // (B, L, 256) bf16
    const float* __restrict__ ffw,           // (256, 256)
    const float* __restrict__ ffb,           // (256)
    float* __restrict__ y)                   // (B, 256, L)
{
  const int b  = blockIdx.z;
  const int l0 = blockIdx.y * 128;
  const int n0 = blockIdx.x * 16;

  __shared__ __attribute__((aligned(16))) unsigned short As[128][34];
  __shared__ __attribute__((aligned(16))) unsigned short Bs[32][16];

  const int tid  = threadIdx.x;
  const int lane = tid & 31;
  const int wid  = tid >> 5;
  const int mbase = wid * 16;
  const int Ml   = lane & 15;
  const int koff = (lane < 16) ? 0 : 8;

  v8f acc = {};
  for (int k0 = 0; k0 < 256; k0 += 32) {
    if (k0 + 32 < 256) {
      __builtin_prefetch(
          ao + ((size_t)(b * 1024 + l0 + (tid >> 1)) << 8) + k0 + 32 + (tid & 1) * 16,
          0, 3);
    }
    // A is already bf16 -> straight u32 copies into padded layout.
    for (int it = 0; it < 8; ++it) {
      int lin = it * 256 + tid;              // 0..2047 dword slots
      int m   = lin >> 4;
      int kp2 = (lin & 15) * 2;
      *(unsigned int*)&As[m][kp2] =
          *(const unsigned int*)&ao[((size_t)(b * 1024 + l0 + m) << 8) + k0 + kp2];
    }
    {
      int kk = tid >> 3;                     // 0..31
      int nn = (tid & 7) * 2;                // 0,2,...,14
      const float* wr = &ffw[(k0 + kk) * 256 + n0 + nn];
      *(unsigned int*)&Bs[kk][nn] = f2bf_pk(wr[0], wr[1]);
    }
    __syncthreads();

    BFrag16 a, bb;
    #pragma unroll
    for (int j = 0; j < 16; ++j) {
      int kk = koff + ((j < 8) ? j : (j + 8));
      a.u[j] = As[mbase + Ml][kk];
    }
    #pragma unroll
    for (int t = 0; t < 16; ++t) bb.u[t] = Bs[koff + ((t < 8) ? t : (t + 8))][Ml];
    acc = __builtin_amdgcn_wmma_f32_16x16x32_bf16(
        false, a.v, false, bb.v, (short)0, acc, false, false);
    __syncthreads();
  }

  const int nn   = lane & 15;
  const int mofs = (lane < 16) ? 0 : 8;
  const int o = n0 + nn;
  const float bias = ffb[o];
  #pragma unroll
  for (int r = 0; r < 8; ++r) {
    int l = l0 + mbase + mofs + r;
    y[((size_t)(b * 256 + o) << 10) + l] = acc[r] + bias;
  }
}

// ---------------------------------------------------------------------------
// Launcher.  Workspace layout (bf16 = ushort):
//   q : 8*8*1024*32  (4 MB) | k : +4 MB | v : +4 MB | ao: +4 MB (B, L, 256)
// ---------------------------------------------------------------------------
extern "C" void kernel_launch(void* const* d_in, const int* in_sizes, int n_in,
                              void* d_out, int out_size, void* d_ws, size_t ws_size,
                              hipStream_t stream) {
  const float* x   = (const float*)d_in[0];
  const float* Qw  = (const float*)d_in[1];
  const float* Kw  = (const float*)d_in[2];
  const float* Vw  = (const float*)d_in[3];
  const float* ffw = (const float*)d_in[4];
  const float* ffb = (const float*)d_in[5];
  const float* rb  = (const float*)d_in[6];
  const int*   ri  = (const int*)d_in[7];

  const size_t perTensor = (size_t)8 * 8 * 1024 * 32;   // elements (ushort)
  unsigned short* qws = (unsigned short*)d_ws;
  unsigned short* kws = qws + perTensor;
  unsigned short* vws = kws + perTensor;
  unsigned short* aws = vws + perTensor;

  qkv_proj_kernel<<<dim3(48, 8, 8), 256, 0, stream>>>(x, Qw, Kw, Vw, qws, kws, vws);
  attn_kernel<<<dim3(8, 64), 256, 0, stream>>>(qws, kws, vws, rb, ri, aws);
  ff_kernel<<<dim3(16, 8, 8), 256, 0, stream>>>(aws, ffw, ffb, (float*)d_out);
}
